// GCNModel_13761075216505
// MI455X (gfx1250) — matrix operand order repaired and checked by reading, not compiled, not run on hardware
//
#include <hip/hip_runtime.h>
#include <hip/hip_bf16.h>

typedef __attribute__((ext_vector_type(16))) __bf16 v16bf;
typedef __attribute__((ext_vector_type(8)))  __bf16 v8bf;
typedef __attribute__((ext_vector_type(8)))  float  v8f;

#define HDIM 128   // hidden width (both conv layers output 128)
#define FDIM 64    // input feature width
#define CCLS 2     // classes

union FragBF {
    v16bf v;
    v8bf  h[2];
};

// ---------------- utility kernels ----------------

__global__ void zero_f32_kernel(float* __restrict__ p, long long n) {
    long long i = (long long)blockIdx.x * blockDim.x + threadIdx.x;
    if (i < n) p[i] = 0.0f;
}

__global__ void f32_to_bf16_kernel(const float* __restrict__ in,
                                   __bf16* __restrict__ out, long long n) {
    long long i = (long long)blockIdx.x * blockDim.x + threadIdx.x;
    if (i < n) out[i] = (__bf16)in[i];
}

// Pre-swizzle weight matrix W[K, NCOLS] (row-major f32) into per-lane WMMA
// B-fragment order (bf16): out[((nt*KT + kt)*32 + lane)*16 + p]
//   where row = kt*32 + (lane<16 ? 0 : 8) + (p&7) + (p>>3)*16, col = nt*16 + (lane&15)
__global__ void swizzle_w_kernel(const float* __restrict__ W,
                                 __bf16* __restrict__ out, int K, int NCOLS) {
    int t = blockIdx.x * blockDim.x + threadIdx.x;
    int total = K * NCOLS;
    if (t >= total) return;
    int p    = t & 15;
    int lane = (t >> 4) & 31;
    int ktn  = t >> 9;
    int KT   = K >> 5;
    int kt   = ktn % KT;
    int nt   = ktn / KT;
    int row  = kt * 32 + ((lane < 16) ? 0 : 8) + (p & 7) + ((p >> 3) << 4);
    int col  = nt * 16 + (lane & 15);
    out[t] = (__bf16)W[row * NCOLS + col];
}

// ---------------- degree / normalization ----------------

__global__ void deg_count_kernel(const int* __restrict__ dst,
                                 float* __restrict__ deg, int E) {
    int i = blockIdx.x * blockDim.x + threadIdx.x;
    if (i < E) atomicAdd(&deg[dst[i]], 1.0f);
}

__global__ void dinv_kernel(float* __restrict__ deg, float* __restrict__ dinv, int N) {
    int i = blockIdx.x * blockDim.x + threadIdx.x;
    if (i >= N) return;
    float d = deg[i] + 1.0f;        // self-loop
    deg[i]  = d;
    dinv[i] = rsqrtf(d);
}

// ---------------- WMMA GEMM: C[M,128] = A[M,K](bf16) @ Bsw(K,128 swizzled bf16)
// One wave computes a 16-row x 128-col stripe. KTILES = K/32.

template<int KTILES>
__global__ void gemm_wmma_bf16_kernel(const __bf16* __restrict__ A,
                                      const __bf16* __restrict__ Bsw,
                                      float* __restrict__ C, int M) {
    constexpr int K = KTILES * 32;
    const int lane = threadIdx.x & 31;
    const int tile = blockIdx.x * (blockDim.x >> 5) + (threadIdx.x >> 5);
    const int m0   = tile * 16;
    if (m0 >= M) return;
    const int half = lane >> 4;     // 0: K 0..7/16..23, 1: K 8..15/24..31
    const int r    = lane & 15;

    // A fragments (16-bit A 16x32 layout, ISA 7.12.2): hoisted across n-tiles
    FragBF a[KTILES];
    const __bf16* arow = A + (size_t)(m0 + r) * K + half * 8;
#pragma unroll
    for (int kt = 0; kt < KTILES; ++kt) {
        a[kt].h[0] = *(const v8bf*)(arow + kt * 32);
        a[kt].h[1] = *(const v8bf*)(arow + kt * 32 + 16);
    }

#pragma unroll
    for (int nt = 0; nt < HDIM / 16; ++nt) {
        v8f acc = {};
#pragma unroll
        for (int kt = 0; kt < KTILES; ++kt) {
            FragBF b;
            const __bf16* bp = Bsw + ((size_t)(nt * KTILES + kt) * 32 + lane) * 16;
            b.h[0] = *(const v8bf*)(bp);
            b.h[1] = *(const v8bf*)(bp + 8);
            acc = __builtin_amdgcn_wmma_f32_16x16x32_bf16(
                      false, a[kt].v, false, b.v, (short)0, acc, false, false);
        }
        // C layout: VGPR rr -> row m0 + half*8 + rr, col nt*16 + r  (coalesced 64B rows)
        float* crow = C + (size_t)(m0 + half * 8) * HDIM + nt * 16 + r;
#pragma unroll
        for (int rr = 0; rr < 8; ++rr)
            crow[(size_t)rr * HDIM] = acc[rr];
    }
}

// ---------------- edge scatter-add (L2-resident atomics) ----------------
// thread t -> edge t>>5, feature chunk (t&31)*4

__global__ void edge_agg_kernel(const int* __restrict__ src, const int* __restrict__ dst,
                                const float* __restrict__ dinv,
                                const float* __restrict__ h, float* __restrict__ agg,
                                long long total) {
    long long t = (long long)blockIdx.x * blockDim.x + threadIdx.x;
    if (t >= total) return;
    int e = (int)(t >> 5);
    int f = ((int)t & 31) * 4;
    int s = src[e], d = dst[e];
    float nrm = dinv[s] * dinv[d];
    const float4 hv = *(const float4*)(h + (size_t)s * HDIM + f);
    float* ap = agg + (size_t)d * HDIM + f;
    atomicAdd(ap + 0, hv.x * nrm);
    atomicAdd(ap + 1, hv.y * nrm);
    atomicAdd(ap + 2, hv.z * nrm);
    atomicAdd(ap + 3, hv.w * nrm);
}

// ---------------- fused self-loop + bias + ReLU (+bf16 for next GEMM) ------

__global__ void combine_kernel(const float* __restrict__ hraw, const float* __restrict__ agg,
                               const float* __restrict__ deg, const float* __restrict__ bias,
                               float* __restrict__ out_f32, __bf16* __restrict__ out_bf16,
                               long long total) {
    long long idx = (long long)blockIdx.x * blockDim.x + threadIdx.x;
    if (idx >= total) return;
    int n = (int)(idx >> 7);
    int f = (int)idx & 127;
    float v = agg[idx] + hraw[idx] / deg[n] + bias[f];
    v = fmaxf(v, 0.0f);
    if (out_f32)  out_f32[idx]  = v;
    if (out_bf16) out_bf16[idx] = (__bf16)v;
}

// ---------------- global mean pool ----------------

__global__ void pool_kernel(const float* __restrict__ h, const int* __restrict__ batch,
                            float* __restrict__ pooled, long long total) {
    long long idx = (long long)blockIdx.x * blockDim.x + threadIdx.x;
    if (idx >= total) return;
    int n = (int)(idx >> 7);
    int f = (int)idx & 127;
    atomicAdd(&pooled[(size_t)batch[n] * HDIM + f], h[idx]);
}

__global__ void count_kernel(const int* __restrict__ batch, float* __restrict__ counts, int N) {
    int i = blockIdx.x * blockDim.x + threadIdx.x;
    if (i < N) atomicAdd(&counts[batch[i]], 1.0f);
}

// ---------------- FC + softmax (G=512 graphs, trivially small) -------------

__global__ void fc_softmax_kernel(const float* __restrict__ pooled,
                                  const float* __restrict__ counts,
                                  const float* __restrict__ Wfc, const float* __restrict__ bfc,
                                  float* __restrict__ out, int G) {
    int g = blockIdx.x * blockDim.x + threadIdx.x;
    if (g >= G) return;
    float inv = 1.0f / fmaxf(counts[g], 1.0f);
    float l0 = bfc[0], l1 = bfc[1];
    const float* pr = pooled + (size_t)g * HDIM;
#pragma unroll 4
    for (int f = 0; f < HDIM; ++f) {
        float v = pr[f] * inv;
        l0 += v * Wfc[f * CCLS + 0];
        l1 += v * Wfc[f * CCLS + 1];
    }
    float m  = fmaxf(l0, l1);
    float e0 = __expf(l0 - m), e1 = __expf(l1 - m);
    float s  = e0 + e1;
    out[g * CCLS + 0] = e0 / s;
    out[g * CCLS + 1] = e1 / s;
}

// ---------------- launch ----------------

extern "C" void kernel_launch(void* const* d_in, const int* in_sizes, int n_in,
                              void* d_out, int out_size, void* d_ws, size_t ws_size,
                              hipStream_t stream) {
    const float* x     = (const float*)d_in[0];
    const int*   ei    = (const int*)d_in[1];   // [2, E]
    const int*   batch = (const int*)d_in[2];
    const float* W1    = (const float*)d_in[4];
    const float* b1    = (const float*)d_in[5];
    const float* W2    = (const float*)d_in[6];
    const float* b2    = (const float*)d_in[7];
    const float* Wfc   = (const float*)d_in[8];
    const float* bfc   = (const float*)d_in[9];
    float* out = (float*)d_out;

    const int N = in_sizes[0] / FDIM;
    const int E = in_sizes[1] / 2;
    const int G = out_size / CCLS;
    const int* src  = ei;
    const int* dstp = ei + E;

    // workspace carve-up (~129 MB)
    char* p = (char*)d_ws;
    auto alloc = [&](size_t bytes) {
        char* q = p;
        p += (bytes + 255) & ~((size_t)255);
        return q;
    };
    float*  bufA   = (float*)alloc((size_t)N * HDIM * 4);   // h_raw / h2
    float*  bufB   = (float*)alloc((size_t)N * HDIM * 4);   // agg
    __bf16* bufC   = (__bf16*)alloc((size_t)N * HDIM * 2);  // x_bf16 then h1_bf16
    float*  deg    = (float*)alloc((size_t)N * 4);
    float*  dinv   = (float*)alloc((size_t)N * 4);
    __bf16* W1sw   = (__bf16*)alloc((size_t)FDIM * HDIM * 2);
    __bf16* W2sw   = (__bf16*)alloc((size_t)HDIM * HDIM * 2);
    float*  pooled = (float*)alloc((size_t)G * HDIM * 4);
    float*  counts = (float*)alloc((size_t)G * 4);

    const int TB = 256;
    auto nb = [](long long n, int tb) { return (unsigned)((n + tb - 1) / tb); };
    const long long NH = (long long)N * HDIM;
    const unsigned gemmBlocks = nb((N + 15) / 16, 8);   // 8 waves per block

    // degrees + normalization
    zero_f32_kernel<<<nb(N, TB), TB, 0, stream>>>(deg, N);
    deg_count_kernel<<<nb(E, TB), TB, 0, stream>>>(dstp, deg, E);
    dinv_kernel<<<nb(N, TB), TB, 0, stream>>>(deg, dinv, N);

    // bf16 conversions + weight swizzle
    f32_to_bf16_kernel<<<nb((long long)N * FDIM, TB), TB, 0, stream>>>(x, bufC, (long long)N * FDIM);
    swizzle_w_kernel<<<nb(FDIM * HDIM, TB), TB, 0, stream>>>(W1, W1sw, FDIM, HDIM);
    swizzle_w_kernel<<<nb(HDIM * HDIM, TB), TB, 0, stream>>>(W2, W2sw, HDIM, HDIM);

    // ---- layer 1 ----
    gemm_wmma_bf16_kernel<FDIM / 32><<<gemmBlocks, 256, 0, stream>>>(bufC, W1sw, bufA, N);
    zero_f32_kernel<<<nb(NH, TB), TB, 0, stream>>>(bufB, NH);
    edge_agg_kernel<<<nb((long long)E * 32, TB), TB, 0, stream>>>(src, dstp, dinv, bufA, bufB,
                                                                  (long long)E * 32);
    combine_kernel<<<nb(NH, TB), TB, 0, stream>>>(bufA, bufB, deg, b1, nullptr, bufC, NH);

    // ---- layer 2 ----
    gemm_wmma_bf16_kernel<HDIM / 32><<<gemmBlocks, 256, 0, stream>>>(bufC, W2sw, bufA, N);
    zero_f32_kernel<<<nb(NH, TB), TB, 0, stream>>>(bufB, NH);
    edge_agg_kernel<<<nb((long long)E * 32, TB), TB, 0, stream>>>(src, dstp, dinv, bufA, bufB,
                                                                  (long long)E * 32);
    combine_kernel<<<nb(NH, TB), TB, 0, stream>>>(bufA, bufB, deg, b2, bufA, nullptr, NH);

    // ---- pool + classifier ----
    zero_f32_kernel<<<nb((long long)G * HDIM, TB), TB, 0, stream>>>(pooled, (long long)G * HDIM);
    zero_f32_kernel<<<nb(G, TB), TB, 0, stream>>>(counts, G);
    pool_kernel<<<nb(NH, TB), TB, 0, stream>>>(bufA, batch, pooled, NH);
    count_kernel<<<nb(N, TB), TB, 0, stream>>>(batch, counts, N);
    fc_softmax_kernel<<<nb(G, TB), TB, 0, stream>>>(pooled, counts, Wfc, bfc, out, G);
}
